// QuantumTransformer_19576460935327
// MI455X (gfx1250) — matrix operand order repaired and checked by reading, not compiled, not run on hardware
//
#include <hip/hip_runtime.h>
#include <hip/hip_bf16.h>
#include <math.h>

// Model dims
#define B_   2
#define L_   1024
#define D_   512
#define H_   8
#define NQ_  8
#define DFF_ 2048
#define V_   32000
#define NL_  2
#define HD_  64
#define D4_  128
#define M_   (B_*L_)   // 2048 rows

typedef unsigned short u16;
typedef __attribute__((ext_vector_type(16))) __bf16 v16bf;
typedef __attribute__((ext_vector_type(8)))  float  v8f;

__device__ __forceinline__ u16 f2bf(float f) {
  unsigned u = __float_as_uint(f);
  u += 0x7FFFu + ((u >> 16) & 1u);          // round-to-nearest-even
  return (u16)(u >> 16);
}
__device__ __forceinline__ float bf2f(u16 h) {
  return __uint_as_float(((unsigned)h) << 16);
}

union Frag16 { v16bf v; u16 s[16]; };
union Pack8  { uint4 u; u16 s[8]; };

// Build a v16bf fragment from two contiguous 16-byte chunks (global or LDS).
__device__ __forceinline__ v16bf load_frag(const u16* p0, const u16* p1) {
  Pack8 a, b;
  a.u = *(const uint4*)p0;
  b.u = *(const uint4*)p1;
  Frag16 r;
#pragma unroll
  for (int i = 0; i < 8; ++i) { r.s[i] = a.s[i]; r.s[8 + i] = b.s[i]; }
  return r.v;
}

__device__ __forceinline__ v8f wmma_bf16(v16bf a, v16bf b, v8f c) {
  return __builtin_amdgcn_wmma_f32_16x16x32_bf16(false, a, false, b, (short)0, c,
                                                 false, false);
}

__device__ __forceinline__ float sigmoidf_(float x) { return 1.f / (1.f + __expf(-x)); }

// Async DMA: 16 bytes/lane from global -> LDS, tracked by ASYNCcnt.
// LDS byte address = low 32 bits of the generic pointer (ISA 10.2: flat->LDS
// mapping truncates the address).
__device__ __forceinline__ void async_stage_b128(const u16* gsrc, const u16* ldst) {
  unsigned lo = (unsigned)(size_t)ldst;
  asm volatile("global_load_async_to_lds_b128 %0, %1, off"
               :: "v"(lo), "v"(gsrc) : "memory");
}
__device__ __forceinline__ void wait_async_le1() {
  asm volatile("s_wait_asynccnt 0x1" ::: "memory");
}
__device__ __forceinline__ void wait_async_le0() {
  asm volatile("s_wait_asynccnt 0x0" ::: "memory");
}

// ---------------------------------------------------------------------------
// Embedding: x = token_embed[ids] + pos_embed ; writes f32 + bf16 copies
// ---------------------------------------------------------------------------
__global__ __launch_bounds__(256)
void embed_kernel(const int* __restrict__ ids, const float* __restrict__ tok,
                  const float* __restrict__ pos, float* __restrict__ xf,
                  u16* __restrict__ xb) {
  int gid = blockIdx.x * 256 + threadIdx.x;      // over M_*D_
  int d  = gid % D_;
  int bl = gid / D_;
  int l  = bl % L_;
  int t  = ids[bl];
  float v = tok[(size_t)t * D_ + d] + pos[(size_t)l * D_ + d];
  xf[gid] = v;
  xb[gid] = f2bf(v);
}

// ---------------------------------------------------------------------------
// f32 -> bf16 straight convert (lm_head is already [N,K])
// ---------------------------------------------------------------------------
__global__ __launch_bounds__(256)
void convert_bf16_kernel(const float* __restrict__ w, u16* __restrict__ o, int n) {
  int gid = blockIdx.x * 256 + threadIdx.x;
  if (gid < n) o[gid] = f2bf(w[gid]);
}

// ---------------------------------------------------------------------------
// W[K,N] f32 -> Wt[N,K] bf16 (so WMMA B-fragments are contiguous loads)
// ---------------------------------------------------------------------------
__global__ __launch_bounds__(256)
void transpose_bf16_kernel(const float* __restrict__ w, u16* __restrict__ o,
                           int K, int N) {
  int gid = blockIdx.x * 256 + threadIdx.x;      // over N*K, output-major
  if (gid >= K * N) return;
  int k = gid % K, n = gid / K;
  o[gid] = f2bf(w[(size_t)k * N + n]);
}

// ---------------------------------------------------------------------------
// GEMM: C[M,N] = act(A[M,K](bf16) @ Wt[N,K](bf16)^T-view + bias)
// Block = 8 waves = 128 rows x 64 cols. The 64x32 B-tile (shared by all 8
// waves) is staged into LDS with double-buffered GLOBAL_LOAD_ASYNC_TO_LDS_B128
// (ASYNCcnt), cutting global B traffic 8x. A-fragments stream from global.
// act: 0 none, 1 tanh, 2 sigmoid, 3 exact gelu
// ---------------------------------------------------------------------------
__global__ __launch_bounds__(256)
void gemm_kernel(const u16* __restrict__ A, const u16* __restrict__ Wt,
                 const float* __restrict__ bias, float* __restrict__ Cf,
                 u16* __restrict__ Cb, int M, int N, int K, int act) {
  __shared__ __align__(16) u16 lb[2][64 * 32];   // double-buffered B tile

  int wave = threadIdx.x >> 5;
  int lane = threadIdx.x & 31;
  int hh = lane >> 4, nn = lane & 15;
  int nbase = blockIdx.x * 64;
  int mbase = (blockIdx.y * 8 + wave) * 16;

  v8f acc[4] = {};
  const u16* arow = A + (size_t)(mbase + nn) * K;      // A-frag: m = lane&15

  // cooperative stager: 256 threads x 16B = 4KB tile (64 rows x 32 k)
  int sr = threadIdx.x >> 2;            // B row 0..63
  int sc = (threadIdx.x & 3) * 8;       // k chunk 0/8/16/24
  const u16* gB = Wt + (size_t)(nbase + sr) * K + sc;

  async_stage_b128(gB, &lb[0][sr * 32 + sc]);          // prologue: tile 0

  int it = 0;
  for (int kb = 0; kb < K; kb += 32, it ^= 1) {
    __syncthreads();                                   // readers of lb[it^1] done
    bool more = (kb + 32) < K;
    if (more) async_stage_b128(gB + kb + 32, &lb[it ^ 1][sr * 32 + sc]);
    if (more) wait_async_le1();                        // tile `it` landed (in-order)
    else      wait_async_le0();
    __syncthreads();                                   // all waves' DMA visible

    __builtin_prefetch(arow + kb + 128, 0, 1);         // global_prefetch_b8
    // A layout: regs0-7 k=16h+[0,16) contiguous
    int a0 = kb + 16 * hh, a1 = kb + 16 * hh + 8;
    v16bf a = load_frag(arow + a0, arow + a1);
    // B layout from LDS: row = 16t+nn, regs0-7 k=16h+[0,16) contiguous
    const u16* bb = &lb[it][0];
    int o0 = 16 * hh;
    acc[0] = wmma_bf16(a, load_frag(bb + (nn     ) * 32 + o0, bb + (nn     ) * 32 + o0 + 8), acc[0]);
    acc[1] = wmma_bf16(a, load_frag(bb + (nn + 16) * 32 + o0, bb + (nn + 16) * 32 + o0 + 8), acc[1]);
    acc[2] = wmma_bf16(a, load_frag(bb + (nn + 32) * 32 + o0, bb + (nn + 32) * 32 + o0 + 8), acc[2]);
    acc[3] = wmma_bf16(a, load_frag(bb + (nn + 48) * 32 + o0, bb + (nn + 48) * 32 + o0 + 8), acc[3]);
  }

#pragma unroll
  for (int t = 0; t < 4; ++t) {
    int n0 = nbase + 16 * t + nn;
    float bs = bias ? bias[n0] : 0.f;
#pragma unroll
    for (int r = 0; r < 8; ++r) {                       // C: M = r + 8*h, N = nn
      float v = acc[t][r] + bs;
      if      (act == 1) v = tanhf(v);
      else if (act == 2) v = sigmoidf_(v);
      else if (act == 3) v = 0.5f * v * (1.f + erff(v * 0.70710678118654752f));
      size_t idx = (size_t)(mbase + r + 8 * hh) * N + n0;
      if (Cf) Cf[idx] = v;
      if (Cb) Cb[idx] = f2bf(v);
    }
  }
}

// ---------------------------------------------------------------------------
// Quantum feature map: out[bl,h,j] = tanh(dot(Q[bl, h*64: ], Wf[:,j]) + bfq[j])
// ---------------------------------------------------------------------------
__global__ __launch_bounds__(128)
void qfeat_kernel(const float* __restrict__ Qf, const float* __restrict__ Wf,
                  const float* __restrict__ bfq, float* __restrict__ out) {
  int gid = blockIdx.x * 128 + threadIdx.x;      // over M_*H_*NQ_
  int j  = gid & (NQ_ - 1);
  int h  = (gid >> 3) & (H_ - 1);
  int bl = gid >> 6;
  const float* q = Qf + (size_t)bl * D_ + h * HD_;
  float s = bfq[j];
#pragma unroll 8
  for (int d = 0; d < HD_; ++d) s += q[d] * Wf[d * NQ_ + j];
  out[gid] = tanhf(s);
}

// ---------------------------------------------------------------------------
// Fused flash attention with quantum interference scores.
// One wave (1-wave workgroup) owns 16 query rows of one (b,h).
// scores = (1-mix)*QK^T/8 + mix*(mix*mean_j cos(qf-kf)*ew_j)
// Online softmax; P tile transposed C->A layout through LDS; PV via WMMA.
// ---------------------------------------------------------------------------
__global__ __launch_bounds__(32)
void attention_kernel(const u16* __restrict__ Qb, const u16* __restrict__ Kb,
                      const u16* __restrict__ Vb, const float* __restrict__ qfm,
                      const float* __restrict__ kfm, const float* __restrict__ qw,
                      const float* __restrict__ entp, float* __restrict__ Of,
                      u16* __restrict__ Ob) {
  __shared__ float lqf[16 * NQ_];
  __shared__ __align__(16) u16 lp[16 * 32];

  int lane = threadIdx.x;
  int hh = lane >> 4, nn = lane & 15;
  int qt = blockIdx.x & 63;            // L/16 = 64 q tiles
  int h  = (blockIdx.x >> 6) & (H_ - 1);
  int b  = blockIdx.x >> 9;
  int qbase = qt * 16;

  float mix = sigmoidf_(entp[0]);
  float ew[NQ_];
#pragma unroll
  for (int j = 0; j < NQ_; ++j) ew[j] = sigmoidf_(qw[h * NQ_ + j]);

  { // stage qf[16 rows][8] into LDS
    int row = lane >> 1, off = (lane & 1) * 4;
    const float* src = qfm + ((size_t)(b * L_ + qbase + row) * H_ + h) * NQ_ + off;
#pragma unroll
    for (int i = 0; i < 4; ++i) lqf[row * NQ_ + off + i] = src[i];
  }
  __syncthreads();

  // Q fragments over d=[0,32) and [32,64) (A layout)
  const u16* qrow = Qb + (size_t)(b * L_ + qbase + nn) * D_ + h * HD_;
  v16bf aQ0 = load_frag(qrow + 16 * hh,      qrow + 16 * hh + 8);
  v16bf aQ1 = load_frag(qrow + 32 + 16 * hh, qrow + 32 + 16 * hh + 8);

  v8f acc[4] = {};
  float mrow[8], lrow[8];
#pragma unroll
  for (int r = 0; r < 8; ++r) { mrow[r] = -1e30f; lrow[r] = 0.f; }

  for (int kb = 0; kb < L_; kb += 32) {
    float s[2][8];
#pragma unroll
    for (int t = 0; t < 2; ++t) {
      // K B-fragment: n = key within tile, k = d (B layout: 16h+[0,16) contiguous)
      const u16* krow = Kb + (size_t)(b * L_ + kb + 16 * t + nn) * D_ + h * HD_;
      v8f c = {};
      c = wmma_bf16(aQ0, load_frag(krow + 16 * hh,      krow + 16 * hh + 8),  c);
      c = wmma_bf16(aQ1, load_frag(krow + 32 + 16 * hh, krow + 32 + 16 * hh + 8), c);
      float kfv[NQ_];
      const float* kfs = kfm + ((size_t)(b * L_ + kb + 16 * t + nn) * H_ + h) * NQ_;
#pragma unroll
      for (int j = 0; j < NQ_; ++j) kfv[j] = kfs[j];
#pragma unroll
      for (int r = 0; r < 8; ++r) {
        float qs = 0.f;
#pragma unroll
        for (int j = 0; j < NQ_; ++j)
          qs += __cosf(lqf[(r + 8 * hh) * NQ_ + j] - kfv[j]) * ew[j];
        float qsc = qs * 0.125f * mix;                 // mean over NQ, * mix
        s[t][r] = (1.f - mix) * (c[r] * 0.125f) + mix * qsc;
      }
    }
    // online softmax (row lives in the 16 lanes sharing hh)
    float p[2][8];
#pragma unroll
    for (int r = 0; r < 8; ++r) {
      float rm = fmaxf(s[0][r], s[1][r]);
      for (int m = 1; m < 16; m <<= 1) rm = fmaxf(rm, __shfl_xor(rm, m));
      float mn = fmaxf(mrow[r], rm);
      float alpha = __expf(mrow[r] - mn);
      p[0][r] = __expf(s[0][r] - mn);
      p[1][r] = __expf(s[1][r] - mn);
      float rs = p[0][r] + p[1][r];
      for (int m = 1; m < 16; m <<= 1) rs += __shfl_xor(rs, m);
      lrow[r] = lrow[r] * alpha + rs;
      mrow[r] = mn;
#pragma unroll
      for (int t = 0; t < 4; ++t) acc[t][r] = acc[t][r] * alpha;
    }
    // C-layout P -> LDS (bf16) -> A-layout fragment
    __syncthreads();
#pragma unroll
    for (int r = 0; r < 8; ++r) {
      lp[(r + 8 * hh) * 32 + nn]      = f2bf(p[0][r]);
      lp[(r + 8 * hh) * 32 + 16 + nn] = f2bf(p[1][r]);
    }
    __syncthreads();
    const u16* prow = &lp[nn * 32];
    v16bf aP = load_frag(prow + 16 * hh, prow + 16 * hh + 8);
    // V B-fragments: n = d within 16-chunk; k = 16*hh + j (contiguous rows)
#pragma unroll
    for (int t = 0; t < 4; ++t) {
      Frag16 bv;
#pragma unroll
      for (int j = 0; j < 16; ++j)
        bv.s[j] = Vb[(size_t)(b * L_ + kb + 16 * hh + j) * D_ + h * HD_ + 16 * t + nn];
      acc[t] = wmma_bf16(aP, bv.v, acc[t]);
    }
  }

#pragma unroll
  for (int r = 0; r < 8; ++r) {
    float rc = 1.f / lrow[r];
    int q = qbase + r + 8 * hh;
#pragma unroll
    for (int t = 0; t < 4; ++t) {
      float o = acc[t][r] * rc;
      size_t idx = (size_t)(b * L_ + q) * D_ + h * HD_ + 16 * t + nn;
      Of[idx] = o;
      Ob[idx] = f2bf(o);
    }
  }
}

// ---------------------------------------------------------------------------
// y = LN(x + add*gate) ; add/gate optional. Writes f32 + bf16. In-place safe.
// One 128-thread block per row of D_=512.
// ---------------------------------------------------------------------------
__global__ __launch_bounds__(128)
void resid_ln_kernel(const float* __restrict__ x, const float* __restrict__ add,
                     const float* __restrict__ gate, const float* __restrict__ g,
                     const float* __restrict__ bb, float* __restrict__ yf,
                     u16* __restrict__ yb) {
  int row = blockIdx.x, t = threadIdx.x;
  __shared__ float red[4];
  const size_t base = (size_t)row * D_;
  float v[4];
#pragma unroll
  for (int i = 0; i < 4; ++i) {
    int d = i * 128 + t;
    float a = x[base + d];
    if (add) {
      float ad = add[base + d];
      if (gate) ad *= gate[base + d];
      a += ad;
    }
    v[i] = a;
  }
  float s = v[0] + v[1] + v[2] + v[3];
  for (int m = 1; m < 32; m <<= 1) s += __shfl_xor(s, m);
  if ((t & 31) == 0) red[t >> 5] = s;
  __syncthreads();
  float mean = (red[0] + red[1] + red[2] + red[3]) * (1.f / D_);
  __syncthreads();
  float q = 0.f;
#pragma unroll
  for (int i = 0; i < 4; ++i) { float d0 = v[i] - mean; q += d0 * d0; }
  for (int m = 1; m < 32; m <<= 1) q += __shfl_xor(q, m);
  if ((t & 31) == 0) red[t >> 5] = q;
  __syncthreads();
  float var = (red[0] + red[1] + red[2] + red[3]) * (1.f / D_);
  float rstd = rsqrtf(var + 1e-5f);
#pragma unroll
  for (int i = 0; i < 4; ++i) {
    int d = i * 128 + t;
    float y = (v[i] - mean) * rstd * g[d] + bb[d];
    yf[base + d] = y;
    yb[base + d] = f2bf(y);
  }
}

// ---------------------------------------------------------------------------
extern "C" void kernel_launch(void* const* d_in, const int* in_sizes, int n_in,
                              void* d_out, int out_size, void* d_ws, size_t ws_size,
                              hipStream_t stream) {
  (void)in_sizes; (void)n_in; (void)out_size; (void)ws_size;
  const int*   ids  = (const int*)d_in[0];
  const float* tok  = (const float*)d_in[1];
  const float* pos  = (const float*)d_in[2];
  const float* Wq   = (const float*)d_in[3];
  const float* bq   = (const float*)d_in[4];
  const float* Wk   = (const float*)d_in[5];
  const float* bk   = (const float*)d_in[6];
  const float* Wv   = (const float*)d_in[7];
  const float* bv   = (const float*)d_in[8];
  const float* Wo   = (const float*)d_in[9];
  const float* bo   = (const float*)d_in[10];
  const float* qw   = (const float*)d_in[11];
  const float* ent  = (const float*)d_in[12];
  const float* Wf   = (const float*)d_in[13];
  const float* bfp  = (const float*)d_in[14];
  const float* g1w  = (const float*)d_in[15];
  const float* g1b  = (const float*)d_in[16];
  const float* g2w  = (const float*)d_in[17];
  const float* g2b  = (const float*)d_in[18];
  const float* ln1g = (const float*)d_in[19];
  const float* ln1b = (const float*)d_in[20];
  const float* ff1w = (const float*)d_in[21];
  const float* ff1b = (const float*)d_in[22];
  const float* ff2w = (const float*)d_in[23];
  const float* ff2b = (const float*)d_in[24];
  const float* ln2g = (const float*)d_in[25];
  const float* ln2b = (const float*)d_in[26];
  const float* lnfg = (const float*)d_in[27];
  const float* lnfb = (const float*)d_in[28];
  const float* lmw  = (const float*)d_in[29];

  // workspace bump allocator (256B aligned)
  char* wp = (char*)d_ws;
  auto bump = [&](size_t bytes) -> void* {
    void* r = (void*)wp;
    wp += (bytes + 255) & ~(size_t)255;
    return r;
  };
  float* xf   = (float*)bump((size_t)M_ * D_ * 4);
  u16*   xb   = (u16*)  bump((size_t)M_ * D_ * 2);
  u16*   wt   = (u16*)  bump((size_t)V_ * D_ * 2);   // transposed-weight scratch (max = lm_head)
  float* Qf   = (float*)bump((size_t)M_ * D_ * 4);
  float* Kf   = (float*)bump((size_t)M_ * D_ * 4);
  u16*   Qb2  = (u16*)  bump((size_t)M_ * D_ * 2);
  u16*   Kb2  = (u16*)  bump((size_t)M_ * D_ * 2);
  u16*   Vb2  = (u16*)  bump((size_t)M_ * D_ * 2);
  float* qfm  = (float*)bump((size_t)M_ * H_ * NQ_ * 4);
  float* kfm  = (float*)bump((size_t)M_ * H_ * NQ_ * 4);
  float* Of   = (float*)bump((size_t)M_ * D_ * 4);
  u16*   Ob   = (u16*)  bump((size_t)M_ * D_ * 2);
  float* attf = (float*)bump((size_t)M_ * D_ * 4);
  u16*   g1bb = (u16*)  bump((size_t)M_ * D4_ * 2);
  float* gatef= (float*)bump((size_t)M_ * D_ * 4);
  u16*   h1b  = (u16*)  bump((size_t)M_ * DFF_ * 2);
  float* ff2f = (float*)bump((size_t)M_ * D_ * 4);

  const dim3 blk256(256), blk128(128), blk32(32);
  const dim3 gemmDD(D_ / 64, M_ / 128);

  embed_kernel<<<(M_ * D_) / 256, blk256, 0, stream>>>(ids, tok, pos, xf, xb);

  for (int l = 0; l < NL_; ++l) {
    const float* lWq  = Wq  + (size_t)l * D_ * D_;
    const float* lWk  = Wk  + (size_t)l * D_ * D_;
    const float* lWv  = Wv  + (size_t)l * D_ * D_;
    const float* lWo  = Wo  + (size_t)l * D_ * D_;
    const float* lWf  = Wf  + (size_t)l * HD_ * NQ_;
    const float* lg1w = g1w + (size_t)l * D_ * D4_;
    const float* lg2w = g2w + (size_t)l * D4_ * D_;
    const float* lf1w = ff1w + (size_t)l * D_ * DFF_;
    const float* lf2w = ff2w + (size_t)l * DFF_ * D_;

    // Q, K, V projections (bf16 WMMA GEMMs)
    transpose_bf16_kernel<<<(D_ * D_ + 255) / 256, blk256, 0, stream>>>(lWq, wt, D_, D_);
    gemm_kernel<<<gemmDD, blk256, 0, stream>>>(xb, wt, bq + (size_t)l * D_, Qf, Qb2, M_, D_, D_, 0);
    transpose_bf16_kernel<<<(D_ * D_ + 255) / 256, blk256, 0, stream>>>(lWk, wt, D_, D_);
    gemm_kernel<<<gemmDD, blk256, 0, stream>>>(xb, wt, bk + (size_t)l * D_, Kf, Kb2, M_, D_, D_, 0);
    transpose_bf16_kernel<<<(D_ * D_ + 255) / 256, blk256, 0, stream>>>(lWv, wt, D_, D_);
    gemm_kernel<<<gemmDD, blk256, 0, stream>>>(xb, wt, bv + (size_t)l * D_, nullptr, Vb2, M_, D_, D_, 0);

    // quantum feature maps
    qfeat_kernel<<<(M_ * H_ * NQ_) / 128, blk128, 0, stream>>>(Qf, lWf, bfp + (size_t)l * NQ_, qfm);
    qfeat_kernel<<<(M_ * H_ * NQ_) / 128, blk128, 0, stream>>>(Kf, lWf, bfp + (size_t)l * NQ_, kfm);

    // fused attention (flash-style, quantum-interference scores)
    attention_kernel<<<B_ * H_ * (L_ / 16), blk32, 0, stream>>>(
        Qb2, Kb2, Vb2, qfm, kfm, qw + (size_t)l * H_ * NQ_, ent + l, Of, Ob);

    // output projection
    transpose_bf16_kernel<<<(D_ * D_ + 255) / 256, blk256, 0, stream>>>(lWo, wt, D_, D_);
    gemm_kernel<<<gemmDD, blk256, 0, stream>>>(Ob, wt, bo + (size_t)l * D_, attf, nullptr, M_, D_, D_, 0);

    // gate MLP: sigmoid(tanh(x@g1)@g2)
    transpose_bf16_kernel<<<(D_ * D4_ + 255) / 256, blk256, 0, stream>>>(lg1w, wt, D_, D4_);
    gemm_kernel<<<dim3(D4_ / 64, M_ / 128), blk256, 0, stream>>>(xb, wt, g1b + (size_t)l * D4_,
                                                                 nullptr, g1bb, M_, D4_, D_, 1);
    transpose_bf16_kernel<<<(D4_ * D_ + 255) / 256, blk256, 0, stream>>>(lg2w, wt, D4_, D_);
    gemm_kernel<<<gemmDD, blk256, 0, stream>>>(g1bb, wt, g2b + (size_t)l * D_,
                                               gatef, nullptr, M_, D_, D4_, 2);

    // x = LN(x + att*gate)
    resid_ln_kernel<<<M_, blk128, 0, stream>>>(xf, attf, gatef,
                                               ln1g + (size_t)l * D_, ln1b + (size_t)l * D_, xf, xb);

    // FFN: gelu(x@ff1)@ff2
    transpose_bf16_kernel<<<(D_ * DFF_ + 255) / 256, blk256, 0, stream>>>(lf1w, wt, D_, DFF_);
    gemm_kernel<<<dim3(DFF_ / 64, M_ / 128), blk256, 0, stream>>>(xb, wt, ff1b + (size_t)l * DFF_,
                                                                  nullptr, h1b, M_, DFF_, D_, 3);
    transpose_bf16_kernel<<<(DFF_ * D_ + 255) / 256, blk256, 0, stream>>>(lf2w, wt, DFF_, D_);
    gemm_kernel<<<gemmDD, blk256, 0, stream>>>(h1b, wt, ff2b + (size_t)l * D_,
                                               ff2f, nullptr, M_, D_, DFF_, 0);

    // x = LN(x + ff)
    resid_ln_kernel<<<M_, blk128, 0, stream>>>(xf, ff2f, nullptr,
                                               ln2g + (size_t)l * D_, ln2b + (size_t)l * D_, xf, xb);
  }

  // final LN + LM head (lm_head_w is [V,D] == [N,K] already)
  resid_ln_kernel<<<M_, blk128, 0, stream>>>(xf, nullptr, nullptr, lnfg, lnfb, xf, xb);
  convert_bf16_kernel<<<((int)((size_t)V_ * D_) + 255) / 256, blk256, 0, stream>>>(lmw, wt, V_ * D_);
  gemm_kernel<<<dim3(V_ / 64, M_ / 128), blk256, 0, stream>>>(xb, wt, nullptr,
                                                              (float*)d_out, nullptr, M_, V_, D_, 0);
}
